// DiscreteCodebook_25220047962555
// MI455X (gfx1250) — compile-verified, matrix operand order checked by reading, not run on hardware
//
#include <hip/hip_runtime.h>
#include <hip/hip_bf16.h>

// ---------------- types ----------------
typedef __attribute__((ext_vector_type(8)))  float  v8f;
typedef __attribute__((ext_vector_type(8)))  __bf16 v8bf;
typedef __attribute__((ext_vector_type(16))) __bf16 v16bf;
typedef int v4i __attribute__((vector_size(16)));
typedef __attribute__((address_space(1))) v4i* as1_v4i_p;   // global
typedef __attribute__((address_space(3))) v4i* as3_v4i_p;   // LDS

#define NZ     32768
#define NCODES 4096
#define DDIM   256
#define DECAY  0.99f
#define ONE_M_DECAY 0.01f
#define CCOST  0.25f
#define EPSV   1e-5f

// chunking for the WMMA kernel: 64 codes per chunk, double-buffered in LDS
#define CHUNK_CODES 64
#define NCHUNK (NCODES / CHUNK_CODES)
#define NSUB   (CHUNK_CODES / 16)

#if defined(__has_builtin)
#if __has_builtin(__builtin_amdgcn_global_load_async_to_lds_b128) && \
    __has_builtin(__builtin_amdgcn_s_wait_asynccnt)
#define USE_ASYNC_LDS 1
#endif
#endif

__device__ __forceinline__ unsigned short f2bf_rne(float f) {
    unsigned int u = __float_as_uint(f);
    unsigned int r = u + 0x7fffu + ((u >> 16) & 1u);
    return (unsigned short)(r >> 16);
}

// ---------------- kernel: zero scratch accumulators ----------------
__global__ __launch_bounds__(256) void k_zero(float* p, int n) {
    int i = blockIdx.x * 256 + threadIdx.x;
    if (i < n) p[i] = 0.0f;
}

// ---------------- kernel: convert z f32 -> bf16 (float4 granular) ----------------
__global__ __launch_bounds__(256) void k_cvt_z(const float* __restrict__ z,
                                               unsigned short* __restrict__ zb) {
    int i = blockIdx.x * 256 + threadIdx.x;            // over NZ*DDIM/4
    float4 v = ((const float4*)z)[i];
    ushort4 o;
    o.x = f2bf_rne(v.x); o.y = f2bf_rne(v.y);
    o.z = f2bf_rne(v.z); o.w = f2bf_rne(v.w);
    ((ushort4*)zb)[i] = o;
}

// ---------------- kernel: convert weight f32 -> bf16 + e2 = ||w_k||^2 ----------------
// 8 waves/block, one code row per wave, 512 blocks.
__global__ __launch_bounds__(256) void k_cvt_w_e2(const float* __restrict__ w,
                                                  unsigned short* __restrict__ wb,
                                                  float* __restrict__ e2) {
    int wave = threadIdx.x >> 5, lane = threadIdx.x & 31;
    int row  = blockIdx.x * 8 + wave;
    const float* wr = w + (size_t)row * DDIM;
    unsigned short* wbr = wb + (size_t)row * DDIM;
    float s = 0.0f;
    float4 a = ((const float4*)wr)[lane * 2 + 0];
    float4 b = ((const float4*)wr)[lane * 2 + 1];
    s += a.x*a.x + a.y*a.y + a.z*a.z + a.w*a.w;
    s += b.x*b.x + b.y*b.y + b.z*b.z + b.w*b.w;
    ushort4 oa, ob;
    oa.x=f2bf_rne(a.x); oa.y=f2bf_rne(a.y); oa.z=f2bf_rne(a.z); oa.w=f2bf_rne(a.w);
    ob.x=f2bf_rne(b.x); ob.y=f2bf_rne(b.y); ob.z=f2bf_rne(b.z); ob.w=f2bf_rne(b.w);
    ((ushort4*)wbr)[lane * 2 + 0] = oa;
    ((ushort4*)wbr)[lane * 2 + 1] = ob;
    for (int m = 1; m <= 16; m <<= 1) s += __shfl_xor(s, m, 32);
    if (lane == 0) e2[row] = s;
}

// ---------------- kernel: fused WMMA distance + argmin ----------------
// 256 blocks x 256 threads (8 waves). Block owns 128 z rows; wave owns 16.
// Codebook streams through LDS in 64-code chunks, double-buffered via
// async global->LDS DMA (ASYNCcnt) so the next chunk's copy overlaps WMMA.
__global__ __launch_bounds__(256) void k_wmma_argmin(const unsigned short* __restrict__ zb,
                                                     const unsigned short* __restrict__ wb,
                                                     const float* __restrict__ e2,
                                                     int* __restrict__ idx_out) {
    __shared__ unsigned short lds_w[2][CHUNK_CODES * DDIM];   // 2 x 32 KB

    const int tid  = threadIdx.x;
    const int wave = tid >> 5;
    const int lane = tid & 31;
    const int sub  = lane >> 4;        // 0: K 0-7/16-23, 1: K 8-15/24-31 (ISA 16-bit layout)
    const int r    = lane & 15;

    const int row_base = blockIdx.x * 128 + wave * 16;

    // Preload all A fragments for this wave's 16 z rows (held in VGPRs all kernel).
    v16bf afrag[8];
    const unsigned short* zrow = zb + (size_t)(row_base + r) * DDIM;
#pragma unroll
    for (int c = 0; c < 8; ++c) {
        const int k0 = c * 32;
        v8bf lo = *(const v8bf*)(zrow + k0 + sub * 8);
        v8bf hi = *(const v8bf*)(zrow + k0 + 16 + sub * 8);
        afrag[c] = __builtin_shufflevector(lo, hi, 0,1,2,3,4,5,6,7,8,9,10,11,12,13,14,15);
    }

    float best[8];
    int   bidx[8];
#pragma unroll
    for (int j = 0; j < 8; ++j) { best[j] = 3.4e38f; bidx[j] = 0; }

#if USE_ASYNC_LDS
    // issue per-lane async DMA for one 32 KB chunk: 2048 x b128, 8 per thread
    auto prefetch = [&](int chunk, int buf) {
        const unsigned short* src = wb + (size_t)chunk * CHUNK_CODES * DDIM;
        unsigned short* dst = &lds_w[buf][0];
#pragma unroll
        for (int i = 0; i < 8; ++i) {
            const int q = tid + i * 256;               // 16-byte packet index
            __builtin_amdgcn_global_load_async_to_lds_b128(
                (as1_v4i_p)(src + q * 8),
                (as3_v4i_p)(dst + q * 8),
                0, 0);
        }
    };
    prefetch(0, 0);
    prefetch(1, 1);
#endif

#pragma unroll 1
    for (int chunk = 0; chunk < NCHUNK; ++chunk) {
        const int buf = chunk & 1;
        const int cb  = chunk * CHUNK_CODES;

#if USE_ASYNC_LDS
        // wait until this wave's DMA for `chunk` has landed (next chunk may
        // still be in flight: <= 8 outstanding), then sync all waves.
        if (chunk + 1 < NCHUNK) __builtin_amdgcn_s_wait_asynccnt(8);
        else                    __builtin_amdgcn_s_wait_asynccnt(0);
        __syncthreads();
#else
        __syncthreads();
        {   // synchronous fallback: 32 KB = 2048 uint4, 8 per thread
            const uint4* src = (const uint4*)(wb + (size_t)cb * DDIM);
            uint4* dst = (uint4*)&lds_w[buf][0];
#pragma unroll
            for (int i = 0; i < 8; ++i) dst[tid + i * 256] = src[tid + i * 256];
        }
        __syncthreads();
#endif

        // hoist the chunk's e2 values: 4 scalar loads issued up front whose
        // latency hides under the first subtile's ds-loads + WMMAs
        float e2v[NSUB];
#pragma unroll
        for (int st = 0; st < NSUB; ++st) e2v[st] = e2[cb + st * 16 + r];

#pragma unroll
        for (int st = 0; st < NSUB; ++st) {
            const int code = cb + st * 16 + r;         // C-layout: lane column = code
            const unsigned short* wrow = &lds_w[buf][(st * 16 + r) * DDIM];
            v8f acc = {0.f,0.f,0.f,0.f,0.f,0.f,0.f,0.f};
#pragma unroll
            for (int c = 0; c < 8; ++c) {
                const int k0 = c * 32;
                v8bf lo = *(const v8bf*)(wrow + k0 + sub * 8);
                v8bf hi = *(const v8bf*)(wrow + k0 + 16 + sub * 8);
                v16bf bfrag = __builtin_shufflevector(lo, hi, 0,1,2,3,4,5,6,7,8,9,10,11,12,13,14,15);
                acc = __builtin_amdgcn_wmma_f32_16x16x32_bf16(
                        false, afrag[c], false, bfrag, (short)0, acc, false, false);
            }
#pragma unroll
            for (int j = 0; j < 8; ++j) {
                const float s = e2v[st] - 2.0f * acc[j];   // ||z||^2 constant per row: omit
                if (s < best[j]) { best[j] = s; bidx[j] = code; }
            }
        }

#if USE_ASYNC_LDS
        __syncthreads();                               // all waves done reading buf
        if (chunk + 2 < NCHUNK) prefetch(chunk + 2, buf);
#endif
    }

    // reduce (min,argmin) across the 16 lanes holding the same rows
#pragma unroll
    for (int m = 1; m <= 8; m <<= 1) {
#pragma unroll
        for (int j = 0; j < 8; ++j) {
            float ov = __shfl_xor(best[j], m, 32);
            int   oi = __shfl_xor(bidx[j], m, 32);
            if (ov < best[j] || (ov == best[j] && oi < bidx[j])) { best[j] = ov; bidx[j] = oi; }
        }
    }
    if (r == 0) {                                      // lanes 0 and 16
#pragma unroll
        for (int j = 0; j < 8; ++j) {
            const int row = row_base + sub * 8 + j;    // C-layout: VGPR j = row (sub*8+j)
            idx_out[row] = bidx[j];
        }
    }
}

// ---------------- kernel: gather z_q, STE out, loss, counts/dw segment sums ----------------
// wave per row; 8 waves/block; 4096 blocks.
__global__ __launch_bounds__(256) void k_scatter(const float* __restrict__ z,
                                                 const float* __restrict__ weight,
                                                 const int* __restrict__ idx,
                                                 float* __restrict__ o_zq,
                                                 float* __restrict__ o_idx_f,
                                                 float* __restrict__ dw,
                                                 float* __restrict__ counts,
                                                 float* __restrict__ loss_acc) {
    const int wave = threadIdx.x >> 5, lane = threadIdx.x & 31;
    const int row  = blockIdx.x * 8 + wave;
    const int code = idx[row];
    const float* zr  = z      + (size_t)row  * DDIM;
    const float* wr  = weight + (size_t)code * DDIM;
    float* dwr       = dw     + (size_t)code * DDIM;
    float* outr      = o_zq   + (size_t)row  * DDIM;
    float local = 0.0f;
#pragma unroll
    for (int i = 0; i < 8; ++i) {
        const int d = lane + i * 32;
        const float zv = zr[d];
        const float wv = wr[d];
        outr[d] = wv;                          // z_q_ste value == z_q
        const float diff = zv - wv;
        local += diff * diff;
        atomicAdd(&dwr[d], zv);                // segment_sum(z)
    }
    for (int m = 1; m <= 16; m <<= 1) local += __shfl_xor(local, m, 32);
    if (lane == 0) {
        atomicAdd(loss_acc, local);
        atomicAdd(&counts[code], 1.0f);        // segment_sum(ones)
        o_idx_f[row] = (float)code;
    }
}

// ---------------- kernel: Laplace-smoothed cluster sizes + loss finalize ----------------
__global__ __launch_bounds__(256) void k_final_cs(const float* __restrict__ ema_cs,
                                                  const float* __restrict__ counts,
                                                  const float* __restrict__ loss_acc,
                                                  float* __restrict__ o_cs,
                                                  float* __restrict__ ws_cs,
                                                  float* __restrict__ o_loss) {
    __shared__ float red[256];
    const int t = threadIdx.x;
    float pre[16];
    float s = 0.0f;
#pragma unroll
    for (int i = 0; i < 16; ++i) {
        const int k = t + i * 256;
        const float p = DECAY * ema_cs[k] + ONE_M_DECAY * counts[k];
        pre[i] = p;
        s += p;
    }
    red[t] = s;
    __syncthreads();
    for (int off = 128; off > 0; off >>= 1) {
        if (t < off) red[t] += red[t + off];
        __syncthreads();
    }
    const float n = red[0];
    const float denom = n + (float)NCODES * EPSV;
#pragma unroll
    for (int i = 0; i < 16; ++i) {
        const int k = t + i * 256;
        const float v = (pre[i] + EPSV) / denom * n;
        o_cs[k]  = v;
        ws_cs[k] = v;
    }
    if (t == 0) o_loss[0] = CCOST * loss_acc[0] / ((float)NZ * (float)DDIM);
}

// ---------------- kernel: EMA codebook + normalized weights ----------------
__global__ __launch_bounds__(256) void k_final_w(const float* __restrict__ ema_w,
                                                 const float* __restrict__ dw,
                                                 const float* __restrict__ ws_cs,
                                                 float* __restrict__ o_w,
                                                 float* __restrict__ o_ema) {
    const int row = blockIdx.x;
    const int t = threadIdx.x;
    const size_t k = (size_t)row * DDIM + t;
    const float ne = DECAY * ema_w[k] + ONE_M_DECAY * dw[k];
    o_ema[k] = ne;
    o_w[k]   = ne / ws_cs[row];
}

// ---------------- launch ----------------
extern "C" void kernel_launch(void* const* d_in, const int* in_sizes, int n_in,
                              void* d_out, int out_size, void* d_ws, size_t ws_size,
                              hipStream_t stream) {
    const float* z       = (const float*)d_in[0];   // [32768,256]
    const float* weight  = (const float*)d_in[1];   // [4096,256]
    const float* ema_cs  = (const float*)d_in[2];   // [4096]
    const float* ema_w   = (const float*)d_in[3];   // [4096,256]

    // ---- output layout (flat, return order) ----
    float* out   = (float*)d_out;
    float* o_zq  = out;                              // 32768*256
    float* o_idx = o_zq  + (size_t)NZ * DDIM;        // 32768
    float* o_ls  = o_idx + NZ;                       // 1
    float* o_w   = o_ls  + 1;                        // 4096*256
    float* o_cs  = o_w   + (size_t)NCODES * DDIM;    // 4096
    float* o_ema = o_cs  + NCODES;                   // 4096*256

    // ---- workspace layout ----
    char* p = (char*)d_ws;
    unsigned short* zb = (unsigned short*)p; p += (size_t)NZ * DDIM * 2;       // 16 MB
    unsigned short* wb = (unsigned short*)p; p += (size_t)NCODES * DDIM * 2;   //  2 MB
    float* e2     = (float*)p; p += (size_t)NCODES * 4;
    int*   idx_i  = (int*)p;   p += (size_t)NZ * 4;
    float* counts = (float*)p; p += (size_t)NCODES * 4;
    float* dw     = (float*)p; p += (size_t)NCODES * DDIM * 4;                 //  4 MB
    float* lossa  = (float*)p; p += 16;
    float* ws_cs  = (float*)p; p += (size_t)NCODES * 4;

    // 1) zero counts + dw + loss (contiguous region starting at counts)
    const int nzero = NCODES + NCODES * DDIM + 4;
    k_zero<<<(nzero + 255) / 256, 256, 0, stream>>>(counts, nzero);

    // 2) bf16 conversions + code norms
    k_cvt_z<<<(NZ * DDIM / 4) / 256, 256, 0, stream>>>(z, zb);
    k_cvt_w_e2<<<NCODES / 8, 256, 0, stream>>>(weight, wb, e2);

    // 3) fused WMMA distance + argmin (async double-buffered codebook stream)
    k_wmma_argmin<<<NZ / 128, 256, 0, stream>>>(zb, wb, e2, idx_i);

    // 4) gather / STE / loss / segment sums
    k_scatter<<<NZ / 8, 256, 0, stream>>>(z, weight, idx_i, o_zq, o_idx, dw, counts, lossa);

    // 5) smoothed cluster sizes + loss
    k_final_cs<<<1, 256, 0, stream>>>(ema_cs, counts, lossa, o_cs, ws_cs, o_ls);

    // 6) EMA weights + normalized codebook
    k_final_w<<<NCODES, 256, 0, stream>>>(ema_w, dw, ws_cs, o_w, o_ema);
}
